// get_model_26405458936161
// MI455X (gfx1250) — compile-verified, matrix operand order checked by reading, not run on hardware
//
#include <hip/hip_runtime.h>
#include <hip/hip_bf16.h>
#include <math.h>
#include <stdint.h>

// ---------------------------------------------------------------------------
// EFM forward for MI455X (gfx1250, wave32, WMMA 16x16x32 bf16)
// B=2, N=2048, C=9, K=16
// ---------------------------------------------------------------------------

#define BATCH 2
#define NPTS  2048
#define CCH   9
#define KNN   16

typedef __attribute__((ext_vector_type(16))) __bf16 v16bf;
typedef __attribute__((ext_vector_type(8)))  __bf16 v8bf;
typedef __attribute__((ext_vector_type(8)))  float  v8f;
typedef __attribute__((ext_vector_type(4)))  float  v4f;
typedef int v4i_vs __attribute__((vector_size(16)));   // matches builtin param

// Async global->LDS copy (CDNA5 GLOBAL_LOAD_ASYNC_TO_LDS_B128, ASYNCcnt),
// guarded so the file compiles on toolchains without the builtin.
#if defined(__has_builtin)
#if __has_builtin(__builtin_amdgcn_global_load_async_to_lds_b128) && \
    __has_builtin(__builtin_amdgcn_s_wait_asynccnt)
#define EFM_ASYNC_LDS 1
#endif
#endif
#ifndef EFM_ASYNC_LDS
#define EFM_ASYNC_LDS 0
#endif

__device__ __forceinline__ void copy16_g2l(const float* g, float* l) {
#if EFM_ASYNC_LDS
  __builtin_amdgcn_global_load_async_to_lds_b128((v4i_vs*)g, (v4i_vs*)l, 0, 0);
#else
  *(v4f*)l = *(const v4f*)g;
#endif
}
__device__ __forceinline__ void wait_stage() {
#if EFM_ASYNC_LDS
  __builtin_amdgcn_s_wait_asynccnt(0);
#endif
}

// k-offset inside a 32-wide K block for 16-bit A/B WMMA fragments (wave32):
// lanes 0-15 cover k {0..7,16..23}, lanes 16-31 cover k {8..15,24..31}.
__device__ __forceinline__ int koff(int e, int hi) {
  return (hi ? 8 : 0) + (e & 7) + ((e & 8) ? 16 : 0);
}

__device__ __forceinline__ v8f wmma_bf16(v16bf a, v16bf b, v8f c) {
  return __builtin_amdgcn_wmma_f32_16x16x32_bf16(
      /*neg_a=*/false, a, /*neg_b=*/false, b,
      /*c_mod=*/(short)0, c, /*reuse_a=*/false, /*reuse_b=*/false);
}

// ---------------------------------------------------------------------------
// Weight pre-pack: W[o, wcol0 + k] (f32, row stride ldW) -> Wp[o, k] bf16,
// K zero-padded to a multiple of 32 so GEMM A-fragments are two aligned
// b128 loads with no bounds checks.
// ---------------------------------------------------------------------------
__global__ void pack_w_kernel(const float* __restrict__ W, int ldW, int wcol0,
                              __bf16* __restrict__ Wp, int O, int Cin, int Kpad) {
  long i = (long)blockIdx.x * 256 + threadIdx.x;
  if (i >= (long)O * Kpad) return;
  int o = (int)(i / Kpad), k = (int)(i % Kpad);
  Wp[i] = (k < Cin) ? (__bf16)W[(long)o * ldW + wcol0 + k] : (__bf16)0.f;
}

// Stage one 32(k) x 64(n) f32 tile into LDS (row stride 68 to kill bank
// conflicts); rows past Cin are zero-filled. 128 threads x 4 b128 chunks.
__device__ __forceinline__ void stage_tile(float* dst, const float* __restrict__ Xb,
                                           int ldX, int kb, int Cin, int tid) {
#pragma unroll
  for (int j = 0; j < 4; ++j) {
    const int q = j * 128 + tid;
    const int k = q >> 4;        // 0..31
    const int c = (q & 15) << 2; // 0..60
    float* l = dst + k * 68 + c;
    if (kb + k < Cin) {
      copy16_g2l(Xb + (long)(kb + k) * ldX + c, l);
    } else {
      v4f z = {0.f, 0.f, 0.f, 0.f};
      *(v4f*)l = z;
    }
  }
}

// ---------------------------------------------------------------------------
// Channel-conv GEMM: Y[b, yrow0+o, n] = epi( Wp[o,:] . X[b, xrow0: , n] )
// Block = 128 thr (4 waves) computing a 64(o) x 64(n) tile. X tile staged in
// LDS via async b128 copies, double buffered; each wave owns 16 o-rows and
// sweeps 4 n-subtiles => 4 WMMAs per A-fragment.
// EPI: 0 = +bias, 2 = +bias,bn,relu, 3 = +bias,sigmoid,*Z
// ---------------------------------------------------------------------------
template <int EPI>
__global__ __launch_bounds__(128) void gemm_wmma_kernel(
    const __bf16* __restrict__ Wp, int Kpad,
    const float* __restrict__ X, int ldX, long xbatch, int xrow0,
    const float* __restrict__ bias, const float* __restrict__ bns,
    const float* __restrict__ bnt,
    float* __restrict__ Y, int ldY, long ybatch, int yrow0,
    const float* __restrict__ Z, int O, int Cin, int Npts) {
  __shared__ float xt[2][32][68];
  const int tid = threadIdx.x;
  const int lane = tid & 31, wv = tid >> 5;
  const int rc = lane & 15, hi = lane >> 4;
  const int n0 = blockIdx.x * 64;
  const int o0 = blockIdx.y * 64 + wv * 16;
  const int b = blockIdx.z;
  if (n0 >= Npts) return;
  const float* Xb = X + (long)b * xbatch + (long)xrow0 * ldX + n0;

  v8f acc[4];
#pragma unroll
  for (int nt = 0; nt < 4; ++nt) acc[nt] = (v8f){0.f,0.f,0.f,0.f,0.f,0.f,0.f,0.f};

  const int nk = (Cin + 31) >> 5;
  stage_tile(&xt[0][0][0], Xb, ldX, 0, Cin, tid);
  wait_stage();
  __syncthreads();

  for (int s = 0; s < nk; ++s) {
    const int kb = s << 5;
    if (s + 1 < nk) stage_tile(&xt[(s + 1) & 1][0][0], Xb, ldX, kb + 32, Cin, tid);
    const float (*xb)[68] = xt[s & 1];
    // A fragment: two aligned 16B loads of packed bf16 weights (L2 resident)
    const __bf16* Ar = Wp + (long)(o0 + rc) * Kpad + kb + (hi ? 8 : 0);
    if (s + 1 < nk) __builtin_prefetch(Ar + 32, 0, 1);  // global_prefetch_b8
    v8bf alo = *(const v8bf*)Ar;
    v8bf ahi = *(const v8bf*)(Ar + 16);
    v16bf af = __builtin_shufflevector(alo, ahi, 0, 1, 2, 3, 4, 5, 6, 7,
                                       8, 9, 10, 11, 12, 13, 14, 15);
#pragma unroll
    for (int nt = 0; nt < 4; ++nt) {
      v16bf bfr;
#pragma unroll
      for (int e = 0; e < 16; ++e) bfr[e] = (__bf16)xb[koff(e, hi)][nt * 16 + rc];
      acc[nt] = wmma_bf16(af, bfr, acc[nt]);
    }
    wait_stage();
    __syncthreads();
  }

  float* Yb = Y + (long)b * ybatch + (long)yrow0 * ldY;
#pragma unroll
  for (int nt = 0; nt < 4; ++nt) {
    const int n = n0 + nt * 16 + rc;
    if (n >= Npts) continue;
#pragma unroll
    for (int v = 0; v < 8; ++v) {
      const int o = o0 + v + 8 * hi;
      float val = acc[nt][v] + (bias ? bias[o] : 0.f);
      if (EPI == 2) val = fmaxf(val * bns[o] + bnt[o], 0.f);
      if (EPI == 3) {
        val = 1.f / (1.f + __expf(-val));
        val *= Z[(long)b * ybatch + (long)(yrow0 + o) * ldY + n];
      }
      Yb[(long)o * ldY + n] = val;
    }
  }
}

// ---------------------------------------------------------------------------
// Small FC: y[b,o] = epi( W[o,:] . x[b,:] + bias[o] )   (latency-bound, VALU)
// ---------------------------------------------------------------------------
template <int EPI>
__global__ void fc_kernel(const float* __restrict__ W, const float* __restrict__ x,
                          const float* __restrict__ bias, const float* __restrict__ s,
                          const float* __restrict__ t, float* __restrict__ y, int O, int I) {
  int o = blockIdx.x * blockDim.x + threadIdx.x;
  int b = blockIdx.y;
  if (o >= O) return;
  const float* xb = x + (long)b * I;
  float acc = bias[o];
  for (int i = 0; i < I; ++i) acc += W[(long)o * I + i] * xb[i];
  if (EPI >= 1) acc = acc * s[o] + t[o];
  if (EPI == 2) acc = fmaxf(acc, 0.f);
  y[(long)b * O + o] = acc;
}

// max over N:  y[b,c] = max_n X[b,c,n]
__global__ void maxreduce_kernel(const float* __restrict__ X, float* __restrict__ y,
                                 int C, int Npts) {
  int c = blockIdx.x * blockDim.x + threadIdx.x;
  int b = blockIdx.y;
  if (c >= C) return;
  const float* xb = X + ((long)b * C + c) * Npts;
  float m = -3.4e38f;
  for (int n = 0; n < Npts; ++n) m = fmaxf(m, xb[n]);
  y[(long)b * C + c] = m;
}

// trans = tf3 + eye(3); write both to scratch and to d_out
__global__ void trans_finalize_kernel(const float* __restrict__ tf3, float* __restrict__ trans,
                                      float* __restrict__ out_trans) {
  int i = threadIdx.x, b = blockIdx.x;
  if (i < 9) {
    float v = tf3[b * 9 + i] + ((i == 0 || i == 4 || i == 8) ? 1.f : 0.f);
    trans[b * 9 + i] = v;
    out_trans[b * 9 + i] = v;
  }
}

// xyz_trans = trans @ xyz; x_trans = [xyz_trans ; x[3:9]]
__global__ void transform_kernel(const float* __restrict__ x, const float* __restrict__ trans,
                                 float* __restrict__ xyz_trans, float* __restrict__ out_xyz,
                                 float* __restrict__ x_trans) {
  int n = blockIdx.x * blockDim.x + threadIdx.x;
  int b = blockIdx.y;
  if (n >= NPTS) return;
  const float* xb = x + (long)b * CCH * NPTS;
  float p0 = xb[0 * NPTS + n], p1 = xb[1 * NPTS + n], p2 = xb[2 * NPTS + n];
  const float* T = trans + b * 9;
#pragma unroll
  for (int i = 0; i < 3; ++i) {
    float v = T[i * 3 + 0] * p0 + T[i * 3 + 1] * p1 + T[i * 3 + 2] * p2;
    xyz_trans[((long)b * 3 + i) * NPTS + n] = v;
    out_xyz[((long)b * 3 + i) * NPTS + n] = v;
    x_trans[((long)b * CCH + i) * NPTS + n] = v;
  }
  for (int i = 3; i < CCH; ++i)
    x_trans[((long)b * CCH + i) * NPTS + n] = xb[i * NPTS + n];
}

// brute-force 16-NN on transformed xyz (one thread per point)
__global__ void knn_kernel(const float* __restrict__ xyzT, int* __restrict__ idx) {
  int n = blockIdx.x * blockDim.x + threadIdx.x;
  int b = blockIdx.y;
  if (n >= NPTS) return;
  const float* xb = xyzT + (long)b * 3 * NPTS;
  float px = xb[n], py = xb[NPTS + n], pz = xb[2 * NPTS + n];
  float bd[KNN]; int bi[KNN];
  for (int i = 0; i < KNN; ++i) { bd[i] = 3.4e38f; bi[i] = 0; }
  for (int m = 0; m < NPTS; ++m) {
    float dx = xb[m] - px, dy = xb[NPTS + m] - py, dz = xb[2 * NPTS + m] - pz;
    float d = dx * dx + dy * dy + dz * dz;
    if (d < bd[KNN - 1]) {
      int j = KNN - 1;
      while (j > 0 && bd[j - 1] > d) { bd[j] = bd[j - 1]; bi[j] = bi[j - 1]; --j; }
      bd[j] = d; bi[j] = m;
    }
  }
  for (int i = 0; i < KNN; ++i) idx[((long)b * NPTS + n) * KNN + i] = bi[i];
}

// ---------------------------------------------------------------------------
// Edge kernel: one workgroup (256 thr = 8 waves) per point.
//  h1[16,1024] = relu(bn1(centerC[n] + neighC[idx] + Wxyz.rel))  (LDS bf16)
//  h2[16,128]  = relu(bn2(h1 @ e2w^T))        <- WMMA, 8 waves x 32 k-steps
//  h3[16,64]   = relu(bn3(h2 @ e3w^T))        <- WMMA, 4 waves x 4 k-steps
//  logits via factored s_c1 + s_c2; softmax over K; weighted sum -> local_feat
// ---------------------------------------------------------------------------
__global__ void edge_kernel(const float* __restrict__ centerC, const float* __restrict__ neighC,
                            const float* __restrict__ e1w, const float* __restrict__ e1s,
                            const float* __restrict__ e1t,
                            const float* __restrict__ sC, const float* __restrict__ sN,
                            const float* __restrict__ s1w, const float* __restrict__ s1s,
                            const float* __restrict__ s1t,
                            const float* __restrict__ s2w, const float* __restrict__ s2b,
                            const float* __restrict__ e2w, const float* __restrict__ e2b,
                            const float* __restrict__ e2s, const float* __restrict__ e2t,
                            const float* __restrict__ e3w, const float* __restrict__ e3b,
                            const float* __restrict__ e3s, const float* __restrict__ e3t,
                            const float* __restrict__ xyzT, const int* __restrict__ idx,
                            float* __restrict__ local_feat) {
  __shared__ __bf16 h1s[16 * 1024];   // 32 KB
  __shared__ float  h2s[16 * 128];    // 8 KB
  __shared__ float  h3s[16 * 64];     // 4 KB
  __shared__ float  s1sh[16 * 128];   // 8 KB
  __shared__ float  relS[16 * 3];
  __shared__ int    idxS[16];
  __shared__ float  logS[16];
  __shared__ float  wS[16];

  const int n = blockIdx.x, b = blockIdx.y, tid = threadIdx.x;
  const float* xb = xyzT + (long)b * 3 * NPTS;

  if (tid < 16) {
    int m = idx[((long)b * NPTS + n) * KNN + tid];
    idxS[tid] = m;
    relS[tid * 3 + 0] = xb[m] - xb[n];
    relS[tid * 3 + 1] = xb[NPTS + m] - xb[NPTS + n];
    relS[tid * 3 + 2] = xb[2 * NPTS + m] - xb[2 * NPTS + n];
  }
  __syncthreads();

  // ---- h1 (factored e_c1 + bn1 + relu), stored bf16 in LDS ----
  {
    const float* cC = centerC + (long)b * 1024 * NPTS;
    const float* nC = neighC + (long)b * 1024 * NPTS;
    for (int c = tid; c < 1024; c += 256) {
      float cc = cC[(long)c * NPTS + n];
      float w0 = e1w[(long)c * 1923 + 1920];
      float w1 = e1w[(long)c * 1923 + 1921];
      float w2 = e1w[(long)c * 1923 + 1922];
      float sc = e1s[c], tc = e1t[c];
      const float* nrow = nC + (long)c * NPTS;
#pragma unroll
      for (int e = 0; e < 16; ++e) {
        float v = cc + nrow[idxS[e]] + w0 * relS[e * 3] + w1 * relS[e * 3 + 1] +
                  w2 * relS[e * 3 + 2];
        v = fmaxf(v * sc + tc, 0.f);
        h1s[e * 1024 + c] = (__bf16)v;
      }
    }
  }
  // ---- s1 (factored s_c1 + bn + relu) ----
  if (tid < 128) {
    int c = tid;
    float cc = sC[((long)b * 128 + c) * NPTS + n];
    float w0 = s1w[(long)c * 1923 + 1920];
    float w1 = s1w[(long)c * 1923 + 1921];
    float w2 = s1w[(long)c * 1923 + 1922];
    float sc = s1s[c], tc = s1t[c];
    const float* nrow = sN + ((long)b * 128 + c) * NPTS;
#pragma unroll
    for (int e = 0; e < 16; ++e) {
      float v = cc + nrow[idxS[e]] + w0 * relS[e * 3] + w1 * relS[e * 3 + 1] +
                w2 * relS[e * 3 + 2];
      s1sh[e * 128 + c] = fmaxf(v * sc + tc, 0.f);
    }
  }
  __syncthreads();

  if (tid < 16) {
    float acc = s2b[0];
    for (int c = 0; c < 128; ++c) acc += s2w[c] * s1sh[tid * 128 + c];
    logS[tid] = acc;
  }

  // ---- h2 = relu(bn2(h1 @ e2w^T)) via WMMA: 8 waves, each a 16-ch tile ----
  {
    const int lane = tid & 31, wave = tid >> 5;
    const int rc = lane & 15, hi = lane >> 4;
    v8f acc = {0.f, 0.f, 0.f, 0.f, 0.f, 0.f, 0.f, 0.f};
    for (int kb = 0; kb < 1024; kb += 32) {
      v16bf af, bfr;
#pragma unroll
      for (int e = 0; e < 16; ++e) {
        int k = kb + koff(e, hi);
        af[e] = h1s[rc * 1024 + k];                               // ds loads
        bfr[e] = (__bf16)e2w[(long)(wave * 16 + rc) * 1024 + k];  // B = K x N
      }
      acc = wmma_bf16(af, bfr, acc);
    }
    int ch = wave * 16 + rc;
#pragma unroll
    for (int v = 0; v < 8; ++v) {
      int e = v + 8 * hi;
      float val = acc[v] + e2b[ch];
      val = fmaxf(val * e2s[ch] + e2t[ch], 0.f);
      h2s[e * 128 + ch] = val;
    }
  }
  __syncthreads();

  // ---- h3 = relu(bn3(h2 @ e3w^T)) via WMMA: waves 0..3 ----
  if (tid < 128) {
    const int lane = tid & 31, wave = tid >> 5;
    const int rc = lane & 15, hi = lane >> 4;
    v8f acc = {0.f, 0.f, 0.f, 0.f, 0.f, 0.f, 0.f, 0.f};
    for (int kb = 0; kb < 128; kb += 32) {
      v16bf af, bfr;
#pragma unroll
      for (int e = 0; e < 16; ++e) {
        int k = kb + koff(e, hi);
        af[e] = (__bf16)h2s[rc * 128 + k];
        bfr[e] = (__bf16)e3w[(long)(wave * 16 + rc) * 128 + k];
      }
      acc = wmma_bf16(af, bfr, acc);
    }
    int ch = wave * 16 + rc;
#pragma unroll
    for (int v = 0; v < 8; ++v) {
      int e = v + 8 * hi;
      float val = acc[v] + e3b[ch];
      val = fmaxf(val * e3s[ch] + e3t[ch], 0.f);
      h3s[e * 64 + ch] = val;
    }
  }
  __syncthreads();

  if (tid == 0) {
    float mx = logS[0];
    for (int e = 1; e < 16; ++e) mx = fmaxf(mx, logS[e]);
    float sum = 0.f;
    for (int e = 0; e < 16; ++e) { float w = __expf(logS[e] - mx); wS[e] = w; sum += w; }
    float inv = 1.f / sum;
    for (int e = 0; e < 16; ++e) wS[e] *= inv;
  }
  __syncthreads();

  if (tid < 64) {
    float acc = 0.f;
    for (int e = 0; e < 16; ++e) acc += h3s[e * 64 + tid] * wS[e];
    local_feat[((long)b * 64 + tid) * NPTS + n] = acc;
  }
}

// row-softmax stats of A[n,m] = xyz_n . xyz_m   (two-pass, one thread / row)
__global__ void rowsoftmax_kernel(const float* __restrict__ xyzT, float* __restrict__ rmax,
                                  float* __restrict__ rsum) {
  int n = blockIdx.x * blockDim.x + threadIdx.x;
  int b = blockIdx.y;
  if (n >= NPTS) return;
  const float* xb = xyzT + (long)b * 3 * NPTS;
  float px = xb[n], py = xb[NPTS + n], pz = xb[2 * NPTS + n];
  float mx = -3.4e38f;
  for (int m = 0; m < NPTS; ++m)
    mx = fmaxf(mx, px * xb[m] + py * xb[NPTS + m] + pz * xb[2 * NPTS + m]);
  float s = 0.f;
  for (int m = 0; m < NPTS; ++m)
    s += __expf(px * xb[m] + py * xb[NPTS + m] + pz * xb[2 * NPTS + m] - mx);
  rmax[(long)b * NPTS + n] = mx;
  rsum[(long)b * NPTS + n] = s;
}

// topo[b,c,m] = sum_n gated[b,c,n] * softmax_row(n)[m]; affinity generated on
// the fly in the B fragment; one wave computes a 64x16 output tile (4 c-tiles
// reuse each B fragment).
__global__ void topo_wmma_kernel(const float* __restrict__ gated, const float* __restrict__ xyzT,
                                 const float* __restrict__ rmax, const float* __restrict__ rsum,
                                 float* __restrict__ topo) {
  const int lane = threadIdx.x;
  const int m0 = blockIdx.x * 16;
  const int b = blockIdx.y;
  const int rc = lane & 15, hi = lane >> 4;
  const float* xb = xyzT + (long)b * 3 * NPTS;
  const int m = m0 + rc;
  const float qx = xb[m], qy = xb[NPTS + m], qz = xb[2 * NPTS + m];
  const float* G = gated + (long)b * 64 * NPTS;
  const float* RM = rmax + (long)b * NPTS;
  const float* RS = rsum + (long)b * NPTS;
  v8f acc[4];
#pragma unroll
  for (int ct = 0; ct < 4; ++ct) acc[ct] = (v8f){0.f,0.f,0.f,0.f,0.f,0.f,0.f,0.f};
  for (int kb = 0; kb < NPTS; kb += 32) {
    v16bf bfr;
#pragma unroll
    for (int e = 0; e < 16; ++e) {
      int kn = kb + koff(e, hi);
      float a = qx * xb[kn] + qy * xb[NPTS + kn] + qz * xb[2 * NPTS + kn];
      bfr[e] = (__bf16)(__expf(a - RM[kn]) / RS[kn]);
    }
#pragma unroll
    for (int ct = 0; ct < 4; ++ct) {
      v16bf af;
#pragma unroll
      for (int e = 0; e < 16; ++e) {
        int kn = kb + koff(e, hi);
        af[e] = (__bf16)G[(long)(ct * 16 + rc) * NPTS + kn];
      }
      acc[ct] = wmma_bf16(af, bfr, acc[ct]);
    }
  }
#pragma unroll
  for (int ct = 0; ct < 4; ++ct)
#pragma unroll
    for (int v = 0; v < 8; ++v) {
      int c = ct * 16 + v + 8 * hi;
      topo[((long)b * 64 + c) * NPTS + m] = acc[ct][v];
    }
}

// ---------------------------------------------------------------------------
// Host orchestration
// ---------------------------------------------------------------------------
extern "C" void kernel_launch(void* const* d_in, const int* in_sizes, int n_in,
                              void* d_out, int out_size, void* d_ws, size_t ws_size,
                              hipStream_t stream) {
  (void)in_sizes; (void)n_in; (void)out_size; (void)ws_size;
  const float* X = (const float*)d_in[0];
  auto P = [&](int i) { return (const float*)d_in[i]; };
  enum {
    T_C1W = 1, T_C1B, T_B1S, T_B1T, T_C2W, T_C2B, T_B2S, T_B2T,
    T_C3W, T_C3B, T_B3S, T_B3T, T_F1W, T_F1B, T_B4S, T_B4T,
    T_F2W, T_F2B, T_B5S, T_B5T, T_F3W, T_F3B,
    M_C1W, M_C1B, M_B1S, M_B1T, M_C2W, M_C2B, M_B2S, M_B2T,
    M_C3W, M_C3B, M_B3S, M_B3T, M_C4W, M_C4B, M_B4S, M_B4T,
    E_C1W, E_C1B, E_B1S, E_B1T, E_C2W, E_C2B, E_B2S, E_B2T,
    E_C3W, E_C3B, E_B3S, E_B3T, S_C1W, S_C1B, S_B1S, S_B1T,
    S_C2W, S_C2B, G_C1W, G_C1B, G_B1S, G_B1T, G_C2W, G_C2B,
    P_C1W, P_C1B, P_B1S, P_B1T
  };

  float* W = (float*)d_ws;
  size_t cur = 0;
  auto alloc = [&](size_t nf) {                 // 16B-aligned f32 scratch
    float* p = W + cur; cur += (nf + 3) & ~(size_t)3; return p;
  };
  auto allocB = [&](size_t ne) {                // 16B-aligned bf16 scratch
    __bf16* p = (__bf16*)(W + cur); cur += (((ne + 1) / 2) + 3) & ~(size_t)3; return p;
  };

  const long BN = (long)BATCH * NPTS;
  float* ht1      = alloc(BN * 64);
  float* ht2      = alloc(BN * 128);
  float* ht3      = alloc(BN * 1024);
  float* tmax     = alloc(BATCH * 1024);
  float* tf1      = alloc(BATCH * 512);
  float* tf2      = alloc(BATCH * 256);
  float* tf3      = alloc(BATCH * 9);
  float* transb   = alloc(BATCH * 9);
  float* xyz_tr   = alloc(BN * 3);
  float* x_tr     = alloc(BN * CCH);
  float* feat_o   = alloc(BN * 960);
  float* feat_t   = alloc(BN * 960);
  float* centerC  = alloc(BN * 1024);
  float* neighC   = alloc(BN * 1024);
  float* sCb      = alloc(BN * 128);
  float* sNb      = alloc(BN * 128);
  int*   idxb     = (int*)alloc(BN * KNN);
  float* localf   = alloc(BN * 64);
  float* gate1    = alloc(BN * 64);
  float* gated    = alloc(BN * 64);
  float* rmax     = alloc(BN);
  float* rsum     = alloc(BN);
  float* topo_raw = alloc(BN * 64);

  // ---- pre-pack conv weights to bf16 (K padded to 32) ----
  auto kpad = [](int c) { return (c + 31) & ~31; };
  auto pack = [&](int pi, int ldW, int wcol0, int O, int Cin) {
    int Kp = kpad(Cin);
    __bf16* dst = allocB((size_t)O * Kp);
    long tot = (long)O * Kp;
    pack_w_kernel<<<dim3((unsigned)((tot + 255) / 256)), 256, 0, stream>>>(
        P(pi), ldW, wcol0, dst, O, Cin, Kp);
    return dst;
  };
  __bf16* wp_tc1 = pack(T_C1W, 3, 0, 64, 3);
  __bf16* wp_tc2 = pack(T_C2W, 64, 0, 128, 64);
  __bf16* wp_tc3 = pack(T_C3W, 128, 0, 1024, 128);
  __bf16* wp_mc1 = pack(M_C1W, CCH, 0, 64, CCH);
  __bf16* wp_mc2 = pack(M_C2W, 64, 0, 128, 64);
  __bf16* wp_mc3 = pack(M_C3W, 128, 0, 256, 128);
  __bf16* wp_mc4 = pack(M_C4W, 256, 0, 512, 256);
  __bf16* wp_e1c = pack(E_C1W, 1923, 0, 1024, 960);
  __bf16* wp_e1n = pack(E_C1W, 1923, 960, 1024, 960);
  __bf16* wp_s1c = pack(S_C1W, 1923, 0, 128, 960);
  __bf16* wp_s1n = pack(S_C1W, 1923, 960, 128, 960);
  __bf16* wp_gc1 = pack(G_C1W, 64, 0, 64, 64);
  __bf16* wp_gc2 = pack(G_C2W, 64, 0, 64, 64);
  __bf16* wp_pc1 = pack(P_C1W, 64, 0, 64, 64);

  float* out       = (float*)d_out;
  float* out_xyz   = out;                                  // [B,3,N]
  float* out_topo  = out + (long)BATCH * 3 * NPTS;         // [B,64,N]
  float* out_trans = out_topo + (long)BATCH * 64 * NPTS;   // [B,3,3]

  auto ggrid = [&](int O) { return dim3(NPTS / 64, O / 64, BATCH); };

  // ---- T-Net ----
  gemm_wmma_kernel<2><<<ggrid(64), 128, 0, stream>>>(
      wp_tc1, kpad(3), X, NPTS, (long)CCH * NPTS, 0, P(T_C1B), P(T_B1S), P(T_B1T),
      ht1, NPTS, (long)64 * NPTS, 0, nullptr, 64, 3, NPTS);
  gemm_wmma_kernel<2><<<ggrid(128), 128, 0, stream>>>(
      wp_tc2, 64, ht1, NPTS, (long)64 * NPTS, 0, P(T_C2B), P(T_B2S), P(T_B2T),
      ht2, NPTS, (long)128 * NPTS, 0, nullptr, 128, 64, NPTS);
  gemm_wmma_kernel<2><<<ggrid(1024), 128, 0, stream>>>(
      wp_tc3, 128, ht2, NPTS, (long)128 * NPTS, 0, P(T_C3B), P(T_B3S), P(T_B3T),
      ht3, NPTS, (long)1024 * NPTS, 0, nullptr, 1024, 128, NPTS);
  maxreduce_kernel<<<dim3(4, BATCH), 256, 0, stream>>>(ht3, tmax, 1024, NPTS);
  fc_kernel<2><<<dim3(2, BATCH), 256, 0, stream>>>(P(T_F1W), tmax, P(T_F1B), P(T_B4S), P(T_B4T), tf1, 512, 1024);
  fc_kernel<2><<<dim3(1, BATCH), 256, 0, stream>>>(P(T_F2W), tf1, P(T_F2B), P(T_B5S), P(T_B5T), tf2, 256, 512);
  fc_kernel<0><<<dim3(1, BATCH), 64, 0, stream>>>(P(T_F3W), tf2, P(T_F3B), nullptr, nullptr, tf3, 9, 256);
  trans_finalize_kernel<<<BATCH, 16, 0, stream>>>(tf3, transb, out_trans);
  transform_kernel<<<dim3(NPTS / 256, BATCH), 256, 0, stream>>>(X, transb, xyz_tr, out_xyz, x_tr);

  // ---- multilevel features (orig and transformed) into [B,960,N] ----
  auto multilevel = [&](const float* inx, float* feat) {
    gemm_wmma_kernel<2><<<ggrid(64), 128, 0, stream>>>(
        wp_mc1, kpad(CCH), inx, NPTS, (long)CCH * NPTS, 0, P(M_C1B), P(M_B1S), P(M_B1T),
        feat, NPTS, (long)960 * NPTS, 0, nullptr, 64, CCH, NPTS);
    gemm_wmma_kernel<2><<<ggrid(128), 128, 0, stream>>>(
        wp_mc2, 64, feat, NPTS, (long)960 * NPTS, 0, P(M_C2B), P(M_B2S), P(M_B2T),
        feat, NPTS, (long)960 * NPTS, 64, nullptr, 128, 64, NPTS);
    gemm_wmma_kernel<2><<<ggrid(256), 128, 0, stream>>>(
        wp_mc3, 128, feat, NPTS, (long)960 * NPTS, 64, P(M_C3B), P(M_B3S), P(M_B3T),
        feat, NPTS, (long)960 * NPTS, 192, nullptr, 256, 128, NPTS);
    gemm_wmma_kernel<2><<<ggrid(512), 128, 0, stream>>>(
        wp_mc4, 256, feat, NPTS, (long)960 * NPTS, 192, P(M_C4B), P(M_B4S), P(M_B4T),
        feat, NPTS, (long)960 * NPTS, 448, nullptr, 512, 256, NPTS);
  };
  multilevel(X, feat_o);
  multilevel(x_tr, feat_t);

  // ---- factored e_c1 / s_c1 per-point GEMMs ----
  gemm_wmma_kernel<0><<<ggrid(1024), 128, 0, stream>>>(
      wp_e1c, 960, feat_o, NPTS, (long)960 * NPTS, 0, P(E_C1B), nullptr, nullptr,
      centerC, NPTS, (long)1024 * NPTS, 0, nullptr, 1024, 960, NPTS);
  gemm_wmma_kernel<0><<<ggrid(1024), 128, 0, stream>>>(
      wp_e1n, 960, feat_t, NPTS, (long)960 * NPTS, 0, nullptr, nullptr, nullptr,
      neighC, NPTS, (long)1024 * NPTS, 0, nullptr, 1024, 960, NPTS);
  gemm_wmma_kernel<0><<<ggrid(128), 128, 0, stream>>>(
      wp_s1c, 960, feat_o, NPTS, (long)960 * NPTS, 0, P(S_C1B), nullptr, nullptr,
      sCb, NPTS, (long)128 * NPTS, 0, nullptr, 128, 960, NPTS);
  gemm_wmma_kernel<0><<<ggrid(128), 128, 0, stream>>>(
      wp_s1n, 960, feat_t, NPTS, (long)960 * NPTS, 0, nullptr, nullptr, nullptr,
      sNb, NPTS, (long)128 * NPTS, 0, nullptr, 128, 960, NPTS);

  // ---- KNN + per-point edge MLP / attention ----
  knn_kernel<<<dim3(NPTS / 256, BATCH), 256, 0, stream>>>(xyz_tr, idxb);
  edge_kernel<<<dim3(NPTS, BATCH), 256, 0, stream>>>(
      centerC, neighC, P(E_C1W), P(E_B1S), P(E_B1T),
      sCb, sNb, P(S_C1W), P(S_B1S), P(S_B1T), P(S_C2W), P(S_C2B),
      P(E_C2W), P(E_C2B), P(E_B2S), P(E_B2T),
      P(E_C3W), P(E_C3B), P(E_B3S), P(E_B3T),
      xyz_tr, idxb, localf);

  // ---- gate ----
  gemm_wmma_kernel<2><<<ggrid(64), 128, 0, stream>>>(
      wp_gc1, 64, localf, NPTS, (long)64 * NPTS, 0, P(G_C1B), P(G_B1S), P(G_B1T),
      gate1, NPTS, (long)64 * NPTS, 0, nullptr, 64, 64, NPTS);
  gemm_wmma_kernel<3><<<ggrid(64), 128, 0, stream>>>(
      wp_gc2, 64, gate1, NPTS, (long)64 * NPTS, 0, P(G_C2B), nullptr, nullptr,
      gated, NPTS, (long)64 * NPTS, 0, localf, 64, 64, NPTS);

  // ---- affinity softmax (fused) + topo matmul + p_c1 ----
  rowsoftmax_kernel<<<dim3(NPTS / 256, BATCH), 256, 0, stream>>>(xyz_tr, rmax, rsum);
  topo_wmma_kernel<<<dim3(NPTS / 16, BATCH), 32, 0, stream>>>(gated, xyz_tr, rmax, rsum, topo_raw);
  gemm_wmma_kernel<2><<<ggrid(64), 128, 0, stream>>>(
      wp_pc1, 64, topo_raw, NPTS, (long)64 * NPTS, 0, P(P_C1B), P(P_B1S), P(P_B1T),
      out_topo, NPTS, (long)64 * NPTS, 0, nullptr, 64, 64, NPTS);
}
